// MultiHeadAttention_6365141532971
// MI455X (gfx1250) — matrix operand order repaired and checked by reading, not compiled, not run on hardware
//
#include <hip/hip_runtime.h>

// ---------------------------------------------------------------------------
// MultiHeadAttention forward for MI455X (gfx1250), bf16 WMMA pipeline with
// async global->LDS double-buffered GEMMs and a flash-attention core.
// B=2, S=2048, D=2048, H=16, HD=128.
// ---------------------------------------------------------------------------

#define DEV static __device__ __forceinline__

typedef __attribute__((ext_vector_type(16))) __bf16          bf16x16;
typedef __attribute__((ext_vector_type(8)))  __bf16          bf16x8;
typedef __attribute__((ext_vector_type(8)))  float           f32x8;
typedef __attribute__((ext_vector_type(16))) unsigned short  u16x16;

static constexpr int Bb = 2;
static constexpr int Ss = 2048;
static constexpr int Dd = 2048;
static constexpr int Hh = 16;
static constexpr int HD = 128;
static constexpr int NT = Bb * Ss;              // 4096 tokens

// LDS tiling for the GEMM: per buffer, A(128x32) and B(128x32) bf16 tiles,
// row pitch 40 elements (80 B = 20 banks -> 16 rows cover all 64 banks once).
static constexpr int LP        = 40;            // LDS row pitch (elements)
static constexpr int TILE_ELE  = 128 * LP;      // 5120 elements per tile
static constexpr unsigned BUF_BYTES = 2u * TILE_ELE * 2u;  // A+B per buffer

DEV unsigned short f32_to_bf16_bits(float f) {
  unsigned int u = __float_as_uint(f);
  u += 0x7FFFu + ((u >> 16) & 1u);              // round-to-nearest-even
  return (unsigned short)(u >> 16);
}

DEV bf16x16 load_frag16(const unsigned short* p0, const unsigned short* p1) {
  bf16x8 lo = *reinterpret_cast<const bf16x8*>(p0);
  bf16x8 hi = *reinterpret_cast<const bf16x8*>(p1);
  return __builtin_shufflevector(lo, hi, 0,1,2,3,4,5,6,7,8,9,10,11,12,13,14,15);
}

DEV f32x8 wmma_bf16(bf16x16 a, bf16x16 b, f32x8 c) {
  // D = A(16x32) * B(32x16) + C, fp32 accumulate
  return __builtin_amdgcn_wmma_f32_16x16x32_bf16(false, a, false, b,
                                                 (short)0, c, false, false);
}

// Async copy of one 16-byte chunk from global to LDS (CDNA5 async path,
// tracked with ASYNCcnt). saddr form: 32-bit per-lane byte offset + SGPR base.
DEV void async_g2l_b128(unsigned ldsByteOff, unsigned gByteOff, const void* base) {
  asm volatile("global_load_async_to_lds_b128 %0, %1, %2"
               :: "v"(ldsByteOff), "v"(gByteOff), "s"(base)
               : "memory");
}

// ---------------------------------------------------------------------------
// fp32 -> bf16 conversion
// ---------------------------------------------------------------------------
__global__ void __launch_bounds__(256)
k_f32_to_bf16(const float* __restrict__ in, unsigned short* __restrict__ out, int n) {
  int i = blockIdx.x * 256 + threadIdx.x;
  if (i < n) out[i] = f32_to_bf16_bits(in[i]);
}

// ---------------------------------------------------------------------------
// C(M x 2048) = X(M x 2048) @ W^T
// 256 threads / 8 waves per block, 128x128 tile per block, 32x64 per wave.
// K staged through LDS in 32-wide double-buffered steps via async-to-LDS DMA.
// mode 0: bf16 out, (b,h,s,hd) scatter        (Q / K)
// mode 1: bf16 out, (b,h,hd,s) scatter        (V transposed)
// mode 2: fp32  out, row-major + bias         (output projection)
// ---------------------------------------------------------------------------
__global__ void __launch_bounds__(256)
k_gemm_xwT(const unsigned short* __restrict__ X,
           const unsigned short* __restrict__ W,
           unsigned short* __restrict__ outb,
           float* __restrict__ outf,
           const float* __restrict__ bias,
           int mode) {
  __shared__ unsigned short smem[2 * 2 * TILE_ELE];   // [buf][A|B][128][LP]

  const int tid    = threadIdx.x;
  const int lane   = tid & 31;
  const int wid    = tid >> 5;
  const int half16 = lane >> 4;
  const int l15    = lane & 15;
  const int m0     = blockIdx.x * 128;
  const int n0     = blockIdx.y * 128;
  const int wrow   = (wid >> 1) * 32;                 // wave 32-row group
  const int wcol   = (wid & 1) * 64;                  // wave 64-col group

  // staging role of this thread: one b128 of A and one of B per k-step
  const int srow = tid >> 1;                          // 0..127
  const int soff = (tid & 1) * 16;                    // 0 or 16 elements
  const unsigned ldsBase = (unsigned)(uintptr_t)smem;
  const unsigned sLds    = (unsigned)((srow * LP + soff) * 2);
  const unsigned aG0     = (unsigned)(((m0 + srow) * Dd + soff) * 2);
  const unsigned bG0     = (unsigned)(((n0 + srow) * Dd + soff) * 2);

  f32x8 acc[2][4] = {};

  // prologue: stage k = 0 into buffer 0
  async_g2l_b128(ldsBase + sLds,                    aG0, X);
  async_g2l_b128(ldsBase + (unsigned)TILE_ELE*2u + sLds, bG0, W);

  int buf = 0;
  for (int k = 0; k < Dd; k += 32) {
    if (k + 32 < Dd) {
      const unsigned nb = (unsigned)(buf ^ 1) * BUF_BYTES;
      async_g2l_b128(ldsBase + nb + sLds,                    aG0 + (unsigned)(k + 32) * 2u, X);
      async_g2l_b128(ldsBase + nb + (unsigned)TILE_ELE*2u + sLds, bG0 + (unsigned)(k + 32) * 2u, W);
      asm volatile("s_wait_asynccnt 2" ::: "memory");   // current buffer landed
    } else {
      asm volatile("s_wait_asynccnt 0" ::: "memory");
    }
    __syncthreads();

    const unsigned short* Abuf = smem + buf * (2 * TILE_ELE);
    const unsigned short* Bbuf = Abuf + TILE_ELE;

    bf16x16 af[2], bfm[4];
#pragma unroll
    for (int mi = 0; mi < 2; ++mi) {
      const unsigned short* p = Abuf + (wrow + mi * 16 + l15) * LP + half16 * 8;
      af[mi] = load_frag16(p, p + 16);
    }
#pragma unroll
    for (int ni = 0; ni < 4; ++ni) {
      const unsigned short* p = Bbuf + (wcol + ni * 16 + l15) * LP + half16 * 8;
      bfm[ni] = load_frag16(p, p + 16);
    }
#pragma unroll
    for (int ni = 0; ni < 4; ++ni)
#pragma unroll
      for (int mi = 0; mi < 2; ++mi)
        acc[mi][ni] = wmma_bf16(af[mi], bfm[ni], acc[mi][ni]);

    __syncthreads();                                   // reads done before overwrite
    buf ^= 1;
  }

  // epilogue: scatter per mode
#pragma unroll
  for (int mi = 0; mi < 2; ++mi) {
#pragma unroll
    for (int ni = 0; ni < 4; ++ni) {
#pragma unroll
      for (int r = 0; r < 8; ++r) {
        const int m = m0 + wrow + mi * 16 + r + half16 * 8;
        const int n = n0 + wcol + ni * 16 + l15;
        const float v = acc[mi][ni][r];
        if (mode == 2) {
          outf[m * Dd + n] = v + bias[n];
        } else {
          const int b  = m >> 11, s  = m & 2047;       // S = 2048
          const int h  = n >> 7,  hd = n & 127;        // HD = 128
          const int idx = (mode == 0)
              ? (((b * Hh + h) * Ss + s) * HD + hd)    // (b,h,s,hd)
              : (((b * Hh + h) * HD + hd) * Ss + s);   // (b,h,hd,s)
          outb[idx] = f32_to_bf16_bits(v);
        }
      }
    }
  }
}

// ---------------------------------------------------------------------------
// Flash attention, one wave per (head, 16 query rows).
// Scores computed transposed (S^T = K_tile x Q^T) so softmax is in-lane and
// the P fragment for P@V needs no cross-lane transpose.
// ---------------------------------------------------------------------------
__global__ void __launch_bounds__(32)
k_attn(const unsigned short* __restrict__ Q,   // (B*H, S, HD)
       const unsigned short* __restrict__ K,   // (B*H, S, HD)
       const unsigned short* __restrict__ Vt,  // (B*H, HD, S)
       unsigned short* __restrict__ ctx)       // (B*S, D) bf16
{
  const int lane   = threadIdx.x;
  const int half16 = lane >> 4;
  const int l15    = lane & 15;
  const int q0     = blockIdx.x * 16;
  const int bh     = blockIdx.y;
  const int b      = bh >> 4;
  const int h      = bh & 15;
  const float scale = 0.08838834764831845f;    // 1/sqrt(128)

  const unsigned short* Qb = Q  + bh * Ss * HD;
  const unsigned short* Kb = K  + bh * Ss * HD;
  const unsigned short* Vb = Vt + bh * HD * Ss;

  // Preload Q as B-fragments (Q^T: lane = query row, dims 8h..8h+7 and 16+8h..16+8h+7).
  bf16x16 qf[4];
  {
    const unsigned short* qRow = Qb + (q0 + l15) * HD + half16 * 8;
#pragma unroll
    for (int kk = 0; kk < 4; ++kk)
      qf[kk] = load_frag16(qRow + kk * 32, qRow + kk * 32 + 16);
  }

  f32x8 o[8] = {};
  float m_run = -__builtin_inff();
  float l_run = 0.f;
  const int qi  = q0 + l15;                    // this lane's query row
  const int aLo = half16 * 8;

  const int nchunk = (q0 + 16 + 31) >> 5;      // causal: keys 0..q0+15
  for (int c = 0; c < nchunk; ++c) {
    const int kb = c << 5;                     // 32-key chunk

    // S^T = K_tile (A, 16 keys x 128) x Q^T (B, 128 x 16 queries)
    f32x8 s0 = {}, s1 = {};
#pragma unroll
    for (int kk = 0; kk < 4; ++kk) {
      const unsigned short* k0 = Kb + (kb + l15) * HD + kk * 32 + aLo;
      const unsigned short* k1 = Kb + (kb + 16 + l15) * HD + kk * 32 + aLo;
      s0 = wmma_bf16(load_frag16(k0, k0 + 16), qf[kk], s0);
      s1 = wmma_bf16(load_frag16(k1, k1 + 16), qf[kk], s1);
    }

    // Preload V fragments now so the vmem clause overlaps the softmax VALU.
    bf16x16 vf[8];
#pragma unroll
    for (int j = 0; j < 8; ++j) {
      const unsigned short* vp = Vb + (j * 16 + l15) * Ss + kb + half16 * 8;
      vf[j] = load_frag16(vp, vp + 16);
    }

    const bool edge = (kb + 31 > q0);          // chunk may cross diagonal
    float mnew = m_run;
#pragma unroll
    for (int r = 0; r < 8; ++r) {
      float v0 = s0[r] * scale;
      float v1 = s1[r] * scale;
      if (edge) {
        if (kb + r + half16 * 8 > qi)      v0 = -__builtin_inff();
        if (kb + 16 + r + half16 * 8 > qi) v1 = -__builtin_inff();
      }
      s0[r] = v0; s1[r] = v1;
      mnew = fmaxf(mnew, fmaxf(v0, v1));
    }
    mnew = fmaxf(mnew, __shfl_xor(mnew, 16));

    const float alpha = __expf(m_run - mnew);
    m_run = mnew;

    float lsum = 0.f;
    u16x16 pu;
#pragma unroll
    for (int r = 0; r < 8; ++r) {
      const float p0 = __expf(s0[r] - mnew);
      const float p1 = __expf(s1[r] - mnew);
      lsum += p0 + p1;
      pu[r]     = f32_to_bf16_bits(p0);        // keys kb+[r|r+8]      (A lo half)
      pu[r + 8] = f32_to_bf16_bits(p1);        // keys kb+16+[r|r+8]   (A hi half)
    }
    lsum  += __shfl_xor(lsum, 16);
    l_run  = l_run * alpha + lsum;

    // Rescale running context: alpha is lane-indexed by query, acc is
    // VGPR-indexed -> broadcast with 8 lane shuffles.
#pragma unroll
    for (int r = 0; r < 8; ++r) {
      const float ar = __shfl(alpha, r + half16 * 8);
#pragma unroll
      for (int j = 0; j < 8; ++j) o[j][r] *= ar;
    }

    // ctx += P (16q x 32k) @ V (32k x 128d), 8 n-subtiles
    const bf16x16 pa = __builtin_bit_cast(bf16x16, pu);
#pragma unroll
    for (int j = 0; j < 8; ++j)
      o[j] = wmma_bf16(pa, vf[j], o[j]);
  }

  // Epilogue: divide by l, scatter to (b, s, h*HD + d) as bf16.
  const float linv = 1.f / l_run;
#pragma unroll
  for (int r = 0; r < 8; ++r) {
    const float lr  = __shfl(linv, r + half16 * 8);
    const int  srow = q0 + r + half16 * 8;
    const int  tok  = b * Ss + srow;
#pragma unroll
    for (int j = 0; j < 8; ++j) {
      const int col = h * HD + j * 16 + l15;
      ctx[tok * Dd + col] = f32_to_bf16_bits(o[j][r] * lr);
    }
  }
}

// ---------------------------------------------------------------------------
// Host-side orchestration (graph-capture safe: launches only).
// ---------------------------------------------------------------------------
extern "C" void kernel_launch(void* const* d_in, const int* in_sizes, int n_in,
                              void* d_out, int out_size, void* d_ws, size_t ws_size,
                              hipStream_t stream) {
  (void)in_sizes; (void)n_in; (void)out_size; (void)ws_size;

  const float* x  = (const float*)d_in[0];
  const float* wq = (const float*)d_in[1];
  const float* wk = (const float*)d_in[2];
  const float* wv = (const float*)d_in[3];
  const float* wo = (const float*)d_in[4];
  const float* bo = (const float*)d_in[5];

  unsigned short* ws  = (unsigned short*)d_ws;
  unsigned short* xb  = ws;                         // NT*Dd
  unsigned short* wqb = xb  + (size_t)NT * Dd;      // Dd*Dd
  unsigned short* wkb = wqb + (size_t)Dd * Dd;
  unsigned short* wvb = wkb + (size_t)Dd * Dd;
  unsigned short* wob = wvb + (size_t)Dd * Dd;
  unsigned short* Qb  = wob + (size_t)Dd * Dd;      // B*H*S*HD
  unsigned short* Kb  = Qb  + (size_t)Bb * Hh * Ss * HD;
  unsigned short* Vt  = Kb  + (size_t)Bb * Hh * Ss * HD;
  unsigned short* ctx = Vt  + (size_t)Bb * Hh * Ss * HD;  // NT*Dd

  const int nx = NT * Dd;
  const int nw = Dd * Dd;
  k_f32_to_bf16<<<(nx + 255) / 256, 256, 0, stream>>>(x,  xb,  nx);
  k_f32_to_bf16<<<(nw + 255) / 256, 256, 0, stream>>>(wq, wqb, nw);
  k_f32_to_bf16<<<(nw + 255) / 256, 256, 0, stream>>>(wk, wkb, nw);
  k_f32_to_bf16<<<(nw + 255) / 256, 256, 0, stream>>>(wv, wvb, nw);
  k_f32_to_bf16<<<(nw + 255) / 256, 256, 0, stream>>>(wo, wob, nw);

  dim3 gg(NT / 128, Dd / 128);   // (32, 16), 8 waves per block
  k_gemm_xwT<<<gg, 256, 0, stream>>>(xb, wqb, Qb, nullptr, nullptr, 0);
  k_gemm_xwT<<<gg, 256, 0, stream>>>(xb, wkb, Kb, nullptr, nullptr, 0);
  k_gemm_xwT<<<gg, 256, 0, stream>>>(xb, wvb, Vt, nullptr, nullptr, 1);

  dim3 ga(Ss / 16, Bb * Hh);     // (128, 32)
  k_attn<<<ga, 32, 0, stream>>>(Qb, Kb, Vt, ctx);

  k_gemm_xwT<<<gg, 256, 0, stream>>>(ctx, wob, nullptr, (float*)d_out, bo, 2);
}
